// WindowAttention_824633721561
// MI455X (gfx1250) — matrix operand order repaired and checked by reading, not compiled, not run on hardware
//
#include <hip/hip_runtime.h>
#include <hip/hip_bf16.h>

typedef __bf16 bf16_t;
typedef bf16_t v16bf __attribute__((ext_vector_type(16)));
typedef float  v8f   __attribute__((ext_vector_type(8)));

#define DIMC    384
#define NTOK    49
#define NHEADS  12
#define HEADDIM 32
#define BWIN    4096
#define NWMASK  64
#define MROWS   (BWIN * NTOK)                             /* 200704 */
#define PLANE   ((size_t)BWIN * NHEADS * NTOK * HEADDIM)  /* 77070336 */

static constexpr float SCALE_F = 0.17677669529663687f;    /* 32^-0.5 */

/* native f32->bf16 (v_cvt_pk_bf16_f32) via compiler conversion */
__device__ __forceinline__ unsigned pack2bf(float x, float y) {
  return (unsigned)__builtin_bit_cast(unsigned short, (bf16_t)x) |
         ((unsigned)__builtin_bit_cast(unsigned short, (bf16_t)y) << 16);
}
__device__ __forceinline__ uint2 pack4bf(float4 f) {
  uint2 r; r.x = pack2bf(f.x, f.y); r.y = pack2bf(f.z, f.w); return r;
}

/* CDNA5 async global->LDS copy (ASYNCcnt path, bypasses VGPRs) */
__device__ __forceinline__ void async_copy_b128(const void* gsrc, void* lds_dst) {
  asm volatile("global_load_async_to_lds_b128 %0, %1, off"
               :: "v"((unsigned)(unsigned long long)lds_dst), "v"(gsrc)
               : "memory");
}
__device__ __forceinline__ void wait_async0() {
  asm volatile("s_wait_asynccnt 0" ::: "memory");
}

/* A fragment (16x32 bf16): half 0 -> K 0..7 & 16..23 ; half 1 -> K 8..15 & 24..31 */
__device__ __forceinline__ v16bf load_fragA(const bf16_t* base, int stride, int row, int k0, int hl) {
  union { v16bf v; uint4 u[2]; } r;
  const bf16_t* p = base + row * stride + k0 + hl * 8;
  r.u[0] = *reinterpret_cast<const uint4*>(p);
  r.u[1] = *reinterpret_cast<const uint4*>(p + 16);
  return r.v;
}
/* B fragment (32x16 bf16): half 0 -> K 0..7 & 16..23 ; half 1 -> K 8..15 & 24..31 */
__device__ __forceinline__ v16bf load_fragB(const bf16_t* base, int stride, int row, int k0, int hl) {
  union { v16bf v; uint4 u[2]; } r;
  const bf16_t* p = base + row * stride + k0 + hl * 8;
  r.u[0] = *reinterpret_cast<const uint4*>(p);
  r.u[1] = *reinterpret_cast<const uint4*>(p + 16);
  return r.v;
}

/* ---------------- Kernel 0: one-shot weight conversion fp32 -> bf16 ---------------- */
__global__ __launch_bounds__(256) void cvt_weights(const float* __restrict__ qw,
                                                   const float* __restrict__ pw,
                                                   bf16_t* __restrict__ qwb,
                                                   bf16_t* __restrict__ pwb) {
  const int NQ = (3 * DIMC * DIMC) / 4;   /* 110592 float4 */
  const int NP = (DIMC * DIMC) / 4;       /* 36864 float4 */
  int i = blockIdx.x * 256 + threadIdx.x;
  if (i < NQ) {
    float4 f = reinterpret_cast<const float4*>(qw)[i];
    reinterpret_cast<uint2*>(qwb)[i] = pack4bf(f);
  } else if (i < NQ + NP) {
    int j = i - NQ;
    float4 f = reinterpret_cast<const float4*>(pw)[j];
    reinterpret_cast<uint2*>(pwb)[j] = pack4bf(f);
  }
}

/* ------------- Kernel 1: QKV GEMM, A-stripe resident, x read exactly once ------------- */
__global__ __launch_bounds__(128) void qkv_gemm(const float* __restrict__ X,
                                                const bf16_t* __restrict__ Wb,
                                                const float* __restrict__ Bq,
                                                bf16_t* __restrict__ qkv_s) {
  __shared__ __align__(16) bf16_t Alds[64 * DIMC];   /* 48 KB */
  __shared__ __align__(16) bf16_t Wlds[64 * DIMC];   /* 48 KB */
  const int t = threadIdx.x;
  const int m0 = blockIdx.x * 64;
  const int wid = t >> 5, lane = t & 31, hl = lane >> 4, lrow = lane & 15;
  const int wm = wid >> 1, wn = wid & 1;

  /* stage A stripe: 64 x 384 fp32 -> bf16 (contiguous 96 KB read) */
  {
    const float* src = X + (size_t)m0 * DIMC;
    for (int i = 0; i < 48; i++) {
      int idx = i * 128 + t;                               /* float4 index */
      float4 f = reinterpret_cast<const float4*>(src)[idx];
      *reinterpret_cast<uint2*>(&Alds[idx * 4]) = pack4bf(f);
    }
  }

  for (int ct = 0; ct < 18; ct++) {
    const int c0 = ct * 64;
    /* async-stage 64x384 bf16 weight tile (contiguous 48 KB, L2-resident) */
    {
      const bf16_t* wsrc = Wb + (size_t)c0 * DIMC;
      for (int i = 0; i < 24; i++) {
        int idx = i * 128 + t;                             /* 16B chunk index */
        async_copy_b128(wsrc + idx * 8, &Wlds[idx * 8]);
      }
    }
    wait_async0();
    __syncthreads();

    v8f acc[2][2] = {};
    for (int k0 = 0; k0 < DIMC; k0 += 32) {
      v16bf a[2], bb[2];
      for (int i = 0; i < 2; i++) a[i]  = load_fragA(Alds, DIMC, wm * 32 + i * 16 + lrow, k0, hl);
      for (int j = 0; j < 2; j++) bb[j] = load_fragB(Wlds, DIMC, wn * 32 + j * 16 + lrow, k0, hl);
      for (int i = 0; i < 2; i++)
        for (int j = 0; j < 2; j++)
          acc[i][j] = __builtin_amdgcn_wmma_f32_16x16x32_bf16(
              false, a[i], false, bb[j], (short)0, acc[i][j], false, false);
    }

    /* scatter into q/k/v planes laid out (b, h, n, d) */
    for (int i = 0; i < 2; i++) {
      for (int j = 0; j < 2; j++) {
        int mbase = m0 + wm * 32 + i * 16 + hl * 8;
        int c = c0 + wn * 32 + j * 16 + lrow;
        float bias = Bq[c];
        int which = c / DIMC;
        int rem = c - which * DIMC;
        int hh = rem >> 5, dd = rem & 31;
        bf16_t* dst = qkv_s + (size_t)which * PLANE;
        for (int r = 0; r < 8; r++) {
          int m = mbase + r;
          int bw = m / NTOK;
          int n = m - bw * NTOK;
          dst[(((size_t)bw * NHEADS + hh) * NTOK + n) * HEADDIM + dd] = (bf16_t)(acc[i][j][r] + bias);
        }
      }
    }
    __syncthreads();
  }
}

/* ------------------- Kernel 2: per-(window,head) attention ------------------- */
__global__ __launch_bounds__(128) void attn_kernel(const bf16_t* __restrict__ Qg,
                                                   const bf16_t* __restrict__ Kg,
                                                   const bf16_t* __restrict__ Vg,
                                                   const float* __restrict__ rpb,
                                                   const float* __restrict__ mask,
                                                   bf16_t* __restrict__ Og) {
  __shared__ __align__(16) bf16_t Qs[64 * 32];
  __shared__ __align__(16) bf16_t Ks[64 * 32];
  __shared__ __align__(16) bf16_t Vt[32 * 64];
  __shared__ __align__(16) float  Sm[64 * 64];
  __shared__ __align__(16) bf16_t Pm[64 * 64];
  const int t = threadIdx.x;
  const int bh = blockIdx.x;
  const int b = bh / NHEADS;
  const int h = bh - b * NHEADS;
  const size_t base = (size_t)bh * NTOK * HEADDIM;

  /* zero-pad rows 49..63 of Q/K tiles */
  for (int idx = t; idx < 240; idx += 128) {
    int row = 49 + (idx >> 4), cp = idx & 15;
    *reinterpret_cast<unsigned*>(&Qs[row * 32 + cp * 2]) = 0u;
    *reinterpret_cast<unsigned*>(&Ks[row * 32 + cp * 2]) = 0u;
  }
  /* async-stage Q,K: 49x32 bf16 = 3136 B contiguous = 196 x 16B chunks */
  for (int idx = t; idx < 196; idx += 128) {
    async_copy_b128(Qg + base + idx * 8, &Qs[idx * 8]);
    async_copy_b128(Kg + base + idx * 8, &Ks[idx * 8]);
  }
  /* V: load + transpose into Vt[d][n] (zero-padded) */
  for (int p = 0; p < 8; p++) {
    int idx = p * 128 + t;
    int row = idx >> 4, cp = idx & 15;
    unsigned uv = 0;
    if (row < NTOK)
      uv = *reinterpret_cast<const unsigned*>(Vg + base + row * 32 + cp * 2);
    int d = cp * 2;
    Vt[d * 64 + row]       = __builtin_bit_cast(bf16_t, (unsigned short)(uv & 0xFFFFu));
    Vt[(d + 1) * 64 + row] = __builtin_bit_cast(bf16_t, (unsigned short)(uv >> 16));
  }
  wait_async0();
  __syncthreads();

  const int wid = t >> 5, lane = t & 31, hl = lane >> 4, lrow = lane & 15;

  /* S = Q K^T  (each wave: one 16-row tile x 4 col tiles, K=32) */
  {
    v16bf a = load_fragA(Qs, 32, wid * 16 + lrow, 0, hl);
    for (int ct = 0; ct < 4; ct++) {
      v16bf bfr = load_fragB(Ks, 32, ct * 16 + lrow, 0, hl);
      v8f c = {};
      c = __builtin_amdgcn_wmma_f32_16x16x32_bf16(false, a, false, bfr, (short)0, c, false, false);
      for (int r = 0; r < 8; r++)
        Sm[(wid * 16 + hl * 8 + r) * 64 + ct * 16 + lrow] = c[r];
    }
  }
  __syncthreads();

  /* softmax with scale + relative-position bias + mask (one thread per row) */
  if (t < 64) {
    int row = t;
    if (row < NTOK) {
      int i1 = row / 7, j1 = row - 7 * i1;
      int w = b & (NWMASK - 1);
      const float* mrow = mask + ((size_t)w * NTOK + row) * NTOK;
      float mx = -1e30f;
      for (int c2 = 0; c2 < NTOK; c2++) {
        int i2 = c2 / 7, j2 = c2 - 7 * i2;
        int rel = (i1 - i2 + 6) * 13 + (j1 - j2 + 6);
        float v = Sm[row * 64 + c2] * SCALE_F + rpb[rel * NHEADS + h] + mrow[c2];
        Sm[row * 64 + c2] = v;
        mx = v > mx ? v : mx;
      }
      float s = 0.f;
      for (int c2 = 0; c2 < NTOK; c2++) {
        float e = __expf(Sm[row * 64 + c2] - mx);
        Sm[row * 64 + c2] = e;
        s += e;
      }
      float inv = 1.0f / s;
      for (int c2 = 0; c2 < NTOK; c2++) Pm[row * 64 + c2] = (bf16_t)(Sm[row * 64 + c2] * inv);
      for (int c2 = NTOK; c2 < 64; c2++) Pm[row * 64 + c2] = __builtin_bit_cast(bf16_t, (unsigned short)0);
    } else {
      for (int c2 = 0; c2 < 64; c2++) Pm[row * 64 + c2] = __builtin_bit_cast(bf16_t, (unsigned short)0);
    }
  }
  __syncthreads();

  /* O = P V  (K=64 -> 2 steps, N=32 -> 2 col tiles) */
  for (int ct = 0; ct < 2; ct++) {
    v8f c = {};
    for (int ks = 0; ks < 2; ks++) {
      v16bf a   = load_fragA(Pm, 64, wid * 16 + lrow, ks * 32, hl);
      v16bf bfr = load_fragB(Vt, 64, ct * 16 + lrow, ks * 32, hl);
      c = __builtin_amdgcn_wmma_f32_16x16x32_bf16(false, a, false, bfr, (short)0, c, false, false);
    }
    for (int r = 0; r < 8; r++) {
      int n = wid * 16 + hl * 8 + r;
      if (n < NTOK)
        Og[((size_t)b * NTOK + n) * DIMC + h * HEADDIM + ct * 16 + lrow] = (bf16_t)c[r];
    }
  }
}

/* ------------- Kernel 3: projection GEMM, A-stripe resident (read once) ------------- */
__global__ __launch_bounds__(128) void proj_gemm(const bf16_t* __restrict__ Ain,
                                                 const bf16_t* __restrict__ Wb,
                                                 const float* __restrict__ Bp,
                                                 float* __restrict__ out) {
  __shared__ __align__(16) bf16_t Alds[64 * DIMC];   /* 48 KB */
  __shared__ __align__(16) bf16_t Wlds[64 * DIMC];   /* 48 KB */
  const int t = threadIdx.x;
  const int m0 = blockIdx.x * 64;
  const int wid = t >> 5, lane = t & 31, hl = lane >> 4, lrow = lane & 15;
  const int wm = wid >> 1, wn = wid & 1;

  /* async-stage A stripe: 64x384 bf16 = contiguous 48 KB */
  {
    const bf16_t* asrc = Ain + (size_t)m0 * DIMC;
    for (int i = 0; i < 24; i++) {
      int idx = i * 128 + t;
      async_copy_b128(asrc + idx * 8, &Alds[idx * 8]);
    }
  }

  for (int ct = 0; ct < 6; ct++) {
    const int c0 = ct * 64;
    {
      const bf16_t* wsrc = Wb + (size_t)c0 * DIMC;
      for (int i = 0; i < 24; i++) {
        int idx = i * 128 + t;
        async_copy_b128(wsrc + idx * 8, &Wlds[idx * 8]);
      }
    }
    wait_async0();
    __syncthreads();

    v8f acc[2][2] = {};
    for (int k0 = 0; k0 < DIMC; k0 += 32) {
      v16bf a[2], bb[2];
      for (int i = 0; i < 2; i++) a[i]  = load_fragA(Alds, DIMC, wm * 32 + i * 16 + lrow, k0, hl);
      for (int j = 0; j < 2; j++) bb[j] = load_fragB(Wlds, DIMC, wn * 32 + j * 16 + lrow, k0, hl);
      for (int i = 0; i < 2; i++)
        for (int j = 0; j < 2; j++)
          acc[i][j] = __builtin_amdgcn_wmma_f32_16x16x32_bf16(
              false, a[i], false, bb[j], (short)0, acc[i][j], false, false);
    }

    for (int i = 0; i < 2; i++) {
      for (int j = 0; j < 2; j++) {
        int mbase = m0 + wm * 32 + i * 16 + hl * 8;
        int c = c0 + wn * 32 + j * 16 + lrow;
        float bias = Bp[c];
        for (int r = 0; r < 8; r++)
          out[(size_t)(mbase + r) * DIMC + c] = acc[i][j][r] + bias;
      }
    }
    __syncthreads();
  }
}

extern "C" void kernel_launch(void* const* d_in, const int* in_sizes, int n_in,
                              void* d_out, int out_size, void* d_ws, size_t ws_size,
                              hipStream_t stream) {
  (void)in_sizes; (void)n_in; (void)out_size; (void)ws_size;
  const float* x      = (const float*)d_in[0];
  const float* mask   = (const float*)d_in[1];
  const float* qkv_w  = (const float*)d_in[2];
  const float* qkv_b  = (const float*)d_in[3];
  const float* proj_w = (const float*)d_in[4];
  const float* proj_b = (const float*)d_in[5];
  const float* rpb    = (const float*)d_in[6];
  float* out = (float*)d_out;

  bf16_t* qkv_s  = (bf16_t*)d_ws;                       /* 3 planes (q,k,v), (b,h,n,d) */
  bf16_t* attn_o = qkv_s + 3 * PLANE;                   /* (b,n,h,d) */
  bf16_t* qwb    = attn_o + PLANE;                      /* qkv_w bf16: 1152x384 */
  bf16_t* pwb    = qwb + (size_t)3 * DIMC * DIMC;       /* proj_w bf16: 384x384 */

  cvt_weights<<<dim3(576), 256, 0, stream>>>(qkv_w, proj_w, qwb, pwb);
  qkv_gemm<<<dim3(MROWS / 64), 128, 0, stream>>>(x, qwb, qkv_b, qkv_s);
  attn_kernel<<<dim3(BWIN * NHEADS), 128, 0, stream>>>(qkv_s, qkv_s + PLANE, qkv_s + 2 * PLANE,
                                                       rpb, mask, attn_o);
  proj_gemm<<<dim3(MROWS / 64), 128, 0, stream>>>(attn_o, pwb, proj_b, out);
}
